// GroupQueryAttention_33827162423863
// MI455X (gfx1250) — matrix operand order repaired and checked by reading, not compiled, not run on hardware
//
#include <hip/hip_runtime.h>

// ---------------------------------------------------------------------------
// CDNA5 (gfx1250) GQA attention: async-LDS double-buffered bf16 WMMA GEMMs
// + async-LDS double-buffered WMMA flash attention.
// ---------------------------------------------------------------------------

typedef __attribute__((ext_vector_type(16))) __bf16 v16bf;
typedef __attribute__((ext_vector_type(8)))  __bf16 v8bf;
typedef __attribute__((ext_vector_type(8)))  float  v8f;

union bfu { v16bf v; v8bf h[2]; };

__device__ __forceinline__ unsigned short f2bf(float f) {
  unsigned u = __builtin_bit_cast(unsigned, f);
  u += 0x7FFFu + ((u >> 16) & 1u);            // round-to-nearest-even
  return (unsigned short)(u >> 16);
}

// async 16B global -> LDS copy (per lane), tracked by ASYNCcnt
__device__ __forceinline__ void async_cp16(unsigned lds_off, const void* gptr) {
  unsigned long long ga = (unsigned long long)(uintptr_t)gptr;
  asm volatile("global_load_async_to_lds_b128 %0, %1, off"
               :: "v"(lds_off), "v"(ga) : "memory");
}

// ---------------------------------------------------------------------------
// fp32 -> bf16 conversion, 4 elements / thread
// ---------------------------------------------------------------------------
__global__ __launch_bounds__(256) void cvt_f32_bf16(const float* __restrict__ in,
                                                    unsigned short* __restrict__ out,
                                                    int n4) {
  int i = blockIdx.x * 256 + threadIdx.x;
  if (i >= n4) return;
  float4 f = reinterpret_cast<const float4*>(in)[i];
  ushort4 r;
  r.x = f2bf(f.x); r.y = f2bf(f.y); r.z = f2bf(f.z); r.w = f2bf(f.w);
  reinterpret_cast<ushort4*>(out)[i] = r;
}

// ---------------------------------------------------------------------------
// Y[M,N] = A[M,K] @ B[N,K]^T + bias[N]   (bf16 in, f32 accumulate)
// Workgroup 256 threads (8 waves), tile 128(m) x 128(n), K-step 64.
// A/B panels staged in LDS via async global->LDS b128 copies, double
// buffered; waves compute 32x64 each (2x4 WMMA fragments) from LDS.
// ---------------------------------------------------------------------------
__global__ __launch_bounds__(256) void gemm_bf16_wmma(
    const unsigned short* __restrict__ A,
    const unsigned short* __restrict__ B,
    const float* __restrict__ bias,
    float* __restrict__ Cf,
    unsigned short* __restrict__ Cb,
    int M, int N, int K, int transOut)
{
  constexpr int KSTEP = 64;
  constexpr int PAD   = 72;            // padded row stride, elements
  constexpr int PANEL = 128 * PAD;     // elements per panel
  __shared__ alignas(16) unsigned short lds[4 * PANEL];  // A0,A1,B0,B1 = 72KB

  const int tid  = threadIdx.x;
  const int lane = tid & 31;
  const int wave = tid >> 5;
  const int wm   = wave & 3;           // 4 waves along m
  const int wn   = wave >> 2;          // 2 waves along n
  const int m0   = blockIdx.y * 128;
  const int n0   = blockIdx.x * 128;
  const int l15  = lane & 15;
  const int aoff = (lane >> 4) << 3;   // A frag: K-offset 0/8  (interleave-8)
  const int boff = (lane >> 4) << 3;   // B frag: K-offset 0/8  (interleave-8)
  const unsigned ldsBase = (unsigned)(uintptr_t)(&lds[0]);

  auto issue = [&](int kc, int d) {
#pragma unroll
    for (int r = 0; r < 4; ++r) {               // A panel: 128 x 64 el = 16KB
      const int c   = r * 256 + tid;            // chunk id 0..1023
      const int row = c >> 3, ck = c & 7;
      const unsigned lo = ldsBase + (unsigned)((d * PANEL + row * PAD + ck * 8) * 2);
      async_cp16(lo, A + (size_t)(m0 + row) * K + kc + ck * 8);
    }
#pragma unroll
    for (int r = 0; r < 4; ++r) {               // B panel
      const int c   = r * 256 + tid;
      const int row = c >> 3, ck = c & 7;
      const unsigned lo = ldsBase + (unsigned)(((2 + d) * PANEL + row * PAD + ck * 8) * 2);
      async_cp16(lo, B + (size_t)(n0 + row) * K + kc + ck * 8);
    }
  };

  v8f acc[2][4] = {};

  issue(0, 0);
  const int NS = K / KSTEP;
  for (int s = 0; s < NS; ++s) {
    const int d = s & 1;
    if (s + 1 < NS) {
      issue((s + 1) * KSTEP, d ^ 1);
      asm volatile("s_wait_asynccnt 0x8" ::: "memory");
    } else {
      asm volatile("s_wait_asynccnt 0x0" ::: "memory");
    }
    __syncthreads();

    const unsigned short* aP = &lds[d * PANEL];
    const unsigned short* bP = &lds[(2 + d) * PANEL];
#pragma unroll
    for (int kk = 0; kk < 2; ++kk) {
      bfu af[2];
#pragma unroll
      for (int mf = 0; mf < 2; ++mf) {
        const unsigned short* p = aP + (wm * 32 + mf * 16 + l15) * PAD + kk * 32 + aoff;
        af[mf].h[0] = *reinterpret_cast<const v8bf*>(p);
        af[mf].h[1] = *reinterpret_cast<const v8bf*>(p + 16);
      }
#pragma unroll
      for (int nf = 0; nf < 4; ++nf) {
        const unsigned short* p = bP + (wn * 64 + nf * 16 + l15) * PAD + kk * 32 + boff;
        bfu bf;
        bf.h[0] = *reinterpret_cast<const v8bf*>(p);
        bf.h[1] = *reinterpret_cast<const v8bf*>(p + 16);
#pragma unroll
        for (int mf = 0; mf < 2; ++mf)
          acc[mf][nf] = __builtin_amdgcn_wmma_f32_16x16x32_bf16(
              false, af[mf].v, false, bf.v, (short)0, acc[mf][nf], false, false);
      }
    }
    __syncthreads();
  }

#pragma unroll
  for (int mf = 0; mf < 2; ++mf) {
    const int rb = m0 + wm * 32 + mf * 16 + ((lane >> 4) << 3);
#pragma unroll
    for (int nf = 0; nf < 4; ++nf) {
      const int cn = n0 + wn * 64 + nf * 16 + l15;
      const float bv = bias ? bias[cn] : 0.0f;
#pragma unroll
      for (int v = 0; v < 8; ++v) {
        const int cm = rb + v;
        const float val = acc[mf][nf][v] + bv;
        if (Cf) Cf[(size_t)cm * N + cn] = val;
        if (Cb) {
          if (transOut) Cb[(size_t)cn * M + cm] = f2bf(val);
          else          Cb[(size_t)cm * N + cn] = f2bf(val);
        }
      }
    }
  }
}

// ---------------------------------------------------------------------------
// Flash attention. Workgroup = 128 threads (4 waves), one head per WG;
// wave w owns q-rows [blockIdx.x*64 + w*16, +16). K/V^T tiles for 64-key
// blocks staged in LDS with async copies, double buffered, shared by all
// 4 waves (4x K/V traffic reduction). Per step per wave: 8 QK^T WMMAs,
// online softmax, P->LDS->A-fragments, 8 PV WMMAs.
// Q: [S,2048] bf16, K: [S,512] bf16, Vt: [512,S] bf16, ctx: [S,2048] bf16.
// ---------------------------------------------------------------------------
__global__ __launch_bounds__(128) void gqa_flash_wmma(
    const unsigned short* __restrict__ Q,
    const unsigned short* __restrict__ K,
    const unsigned short* __restrict__ Vt,
    unsigned short* __restrict__ ctx)
{
  constexpr int S = 2048, DIMS = 2048, HD = 64, KV = 512;
  constexpr int PAD  = 72;
  constexpr int TILE = 64 * PAD;                 // elements per K or V tile
  __shared__ alignas(16) unsigned short kt[2 * TILE];      // K tiles (double)
  __shared__ alignas(16) unsigned short vt[2 * TILE];      // V^T tiles (double)
  __shared__ alignas(16) unsigned short pl[4][16][PAD];    // per-wave P tiles

  const int tid  = threadIdx.x;
  const int lane = tid & 31;
  const int wave = tid >> 5;
  const int h    = blockIdx.y;
  const int g    = h >> 2;                       // rep = H/G = 4, group-major
  const int qr0  = blockIdx.x * 64 + wave * 16;
  const int l15  = lane & 15;
  const int aoff = (lane >> 4) << 3;
  const int boff = (lane >> 4) << 3;
  const float scale = 0.125f;                    // 1/sqrt(64)

  const unsigned short* Kg = K  + g * HD;              // col offset in KV row
  const unsigned short* Vg = Vt + (size_t)(g * HD) * S;
  const unsigned ktBase = (unsigned)(uintptr_t)(&kt[0]);
  const unsigned vtBase = (unsigned)(uintptr_t)(&vt[0]);

  // stage one 64-key block: K tile [key][64d], V^T tile [d][64key]
  auto issue = [&](int kb, int d) {
#pragma unroll
    for (int r = 0; r < 4; ++r) {
      const int c   = r * 128 + tid;             // 0..511
      const int row = c >> 3, ck = c & 7;
      async_cp16(ktBase + (unsigned)((d * TILE + row * PAD + ck * 8) * 2),
                 Kg + (size_t)(kb + row) * KV + ck * 8);
      async_cp16(vtBase + (unsigned)((d * TILE + row * PAD + ck * 8) * 2),
                 Vg + (size_t)row * S + kb + ck * 8);
    }
  };

  // Q fragments (persistent)
  v16bf aq[2];
  {
    const unsigned short* qrow = Q + (size_t)(qr0 + l15) * DIMS + h * HD;
#pragma unroll
    for (int c = 0; c < 2; ++c) {
      bfu u;
      u.h[0] = *reinterpret_cast<const v8bf*>(qrow + c * 32 + aoff);
      u.h[1] = *reinterpret_cast<const v8bf*>(qrow + c * 32 + 16 + aoff);
      aq[c] = u.v;
    }
  }

  v8f o[4] = {};
  float mrow[8], lrow[8];
#pragma unroll
  for (int v = 0; v < 8; ++v) { mrow[v] = -1e30f; lrow[v] = 0.0f; }

  issue(0, 0);
  constexpr int NS = S / 64;
  for (int s = 0; s < NS; ++s) {
    const int d = s & 1;
    if (s + 1 < NS) {
      issue((s + 1) * 64, d ^ 1);
      asm volatile("s_wait_asynccnt 0x8" ::: "memory");
    } else {
      asm volatile("s_wait_asynccnt 0x0" ::: "memory");
    }
    __syncthreads();

    const unsigned short* kP = &kt[d * TILE];
    const unsigned short* vP = &vt[d * TILE];

    // ---- scores: 16 q-rows x 64 keys (4 WMMA column chunks x 2 d-chunks)
    v8f sc[4] = {};
#pragma unroll
    for (int nc = 0; nc < 4; ++nc) {
      const unsigned short* kr = kP + (nc * 16 + l15) * PAD;
#pragma unroll
      for (int c = 0; c < 2; ++c) {
        bfu b;
        b.h[0] = *reinterpret_cast<const v8bf*>(kr + c * 32 + boff);
        b.h[1] = *reinterpret_cast<const v8bf*>(kr + c * 32 + boff + 16);
        sc[nc] = __builtin_amdgcn_wmma_f32_16x16x32_bf16(
            false, aq[c], false, b.v, (short)0, sc[nc], false, false);
      }
    }

    // ---- online softmax over 64 columns
    float alpha[8];
    const int rb = (lane >> 4) << 3;
#pragma unroll
    for (int v = 0; v < 8; ++v) {
      float sv[4];
#pragma unroll
      for (int nc = 0; nc < 4; ++nc) sv[nc] = sc[nc][v] * scale;
      float mx = fmaxf(fmaxf(sv[0], sv[1]), fmaxf(sv[2], sv[3]));
#pragma unroll
      for (int m = 1; m < 16; m <<= 1) mx = fmaxf(mx, __shfl_xor(mx, m, 32));
      const float mn = fmaxf(mrow[v], mx);
      float p[4], ls = 0.0f;
#pragma unroll
      for (int nc = 0; nc < 4; ++nc) { p[nc] = __expf(sv[nc] - mn); ls += p[nc]; }
#pragma unroll
      for (int m = 1; m < 16; m <<= 1) ls += __shfl_xor(ls, m, 32);
      alpha[v] = __expf(mrow[v] - mn);
      lrow[v]  = lrow[v] * alpha[v] + ls;
      mrow[v]  = mn;
#pragma unroll
      for (int nc = 0; nc < 4; ++nc) pl[wave][rb + v][nc * 16 + l15] = f2bf(p[nc]);
    }
#pragma unroll
    for (int c = 0; c < 4; ++c)
#pragma unroll
      for (int v = 0; v < 8; ++v) o[c][v] *= alpha[v];
    asm volatile("" ::: "memory");   // keep P stores before P loads (in-order per wave)

    // ---- P (16x64) as two A fragments
    bfu up[2];
#pragma unroll
    for (int kk = 0; kk < 2; ++kk) {
      up[kk].h[0] = *reinterpret_cast<const v8bf*>(&pl[wave][l15][kk * 32 + aoff]);
      up[kk].h[1] = *reinterpret_cast<const v8bf*>(&pl[wave][l15][kk * 32 + 16 + aoff]);
    }

    // ---- O += P @ V : 4 d-chunks x 2 key-chunks
#pragma unroll
    for (int c = 0; c < 4; ++c) {
      const unsigned short* vr = vP + (c * 16 + l15) * PAD;
#pragma unroll
      for (int kk = 0; kk < 2; ++kk) {
        bfu b;
        b.h[0] = *reinterpret_cast<const v8bf*>(vr + kk * 32 + boff);
        b.h[1] = *reinterpret_cast<const v8bf*>(vr + kk * 32 + boff + 16);
        o[c] = __builtin_amdgcn_wmma_f32_16x16x32_bf16(
            false, up[kk].v, false, b.v, (short)0, o[c], false, false);
      }
    }
    __syncthreads();   // all waves done with buffer d before step s+2 refills it
  }

  const int rb = (lane >> 4) << 3;
#pragma unroll
  for (int c = 0; c < 4; ++c)
#pragma unroll
    for (int v = 0; v < 8; ++v)
      ctx[(size_t)(qr0 + rb + v) * DIMS + h * HD + c * 16 + l15] =
          f2bf(o[c][v] / lrow[v]);
}

// ---------------------------------------------------------------------------
// Launch
// ---------------------------------------------------------------------------
extern "C" void kernel_launch(void* const* d_in, const int* in_sizes, int n_in,
                              void* d_out, int out_size, void* d_ws, size_t ws_size,
                              hipStream_t stream) {
  (void)in_sizes; (void)n_in; (void)out_size; (void)ws_size;
  const float* x  = (const float*)d_in[0];
  const float* wq = (const float*)d_in[1];
  const float* bq = (const float*)d_in[2];
  const float* wk = (const float*)d_in[3];
  const float* bk = (const float*)d_in[4];
  const float* wv = (const float*)d_in[5];
  const float* bv = (const float*)d_in[6];
  const float* wo = (const float*)d_in[7];
  const float* bo = (const float*)d_in[8];
  float* out = (float*)d_out;

  constexpr size_t MB = 1024 * 1024;
  char* ws = (char*)d_ws;
  unsigned short* xb  = (unsigned short*)(ws + 0 * MB);   // [2048,2048] 8MB
  unsigned short* wqb = (unsigned short*)(ws + 8 * MB);   // [2048,2048] 8MB
  unsigned short* wkb = (unsigned short*)(ws + 16 * MB);  // [512,2048]  2MB
  unsigned short* wvb = (unsigned short*)(ws + 18 * MB);  // [512,2048]  2MB
  unsigned short* wob = (unsigned short*)(ws + 20 * MB);  // [2048,2048] 8MB
  unsigned short* Qb  = (unsigned short*)(ws + 28 * MB);  // [2048,2048] 8MB
  unsigned short* Kb  = (unsigned short*)(ws + 36 * MB);  // [2048,512]  2MB
  unsigned short* Vtb = (unsigned short*)(ws + 38 * MB);  // [512,2048]  2MB (V^T)
  unsigned short* ctx = (unsigned short*)(ws + 40 * MB);  // [2048,2048] 8MB

  auto cvt = [&](const float* in, unsigned short* o, int n) {
    int n4 = n / 4;
    cvt_f32_bf16<<<dim3((n4 + 255) / 256), dim3(256), 0, stream>>>(in, o, n4);
  };
  cvt(x,  xb,  2048 * 2048);
  cvt(wq, wqb, 2048 * 2048);
  cvt(wk, wkb, 512 * 2048);
  cvt(wv, wvb, 512 * 2048);
  cvt(wo, wob, 2048 * 2048);

  // Q = x @ wq^T + bq  -> bf16 [2048,2048]
  gemm_bf16_wmma<<<dim3(2048 / 128, 2048 / 128), 256, 0, stream>>>(
      xb, wqb, bq, nullptr, Qb, 2048, 2048, 2048, 0);
  // K = x @ wk^T + bk  -> bf16 [2048,512]
  gemm_bf16_wmma<<<dim3(512 / 128, 2048 / 128), 256, 0, stream>>>(
      xb, wkb, bk, nullptr, Kb, 2048, 512, 2048, 0);
  // V^T = (x @ wv^T + bv)^T -> bf16 [512,2048]
  gemm_bf16_wmma<<<dim3(512 / 128, 2048 / 128), 256, 0, stream>>>(
      xb, wvb, bv, nullptr, Vtb, 2048, 512, 2048, 1);

  // flash attention: 32 q-blocks x 32 heads, 4 waves each
  gqa_flash_wmma<<<dim3(2048 / 64, 32), 128, 0, stream>>>(Qb, Kb, Vtb, ctx);

  // out = ctx @ wo^T + bo -> f32
  gemm_bf16_wmma<<<dim3(2048 / 128, 2048 / 128), 256, 0, stream>>>(
      ctx, wob, bo, out, nullptr, 2048, 2048, 2048, 0);
}
